// SequenceEmbedLoss_48911087566953
// MI455X (gfx1250) — matrix operand order repaired and checked by reading, not compiled, not run on hardware
//
#include <hip/hip_runtime.h>
#include <hip/hip_bf16.h>

typedef float     v8f  __attribute__((ext_vector_type(8)));
typedef _Float16  v16h __attribute__((ext_vector_type(16)));
typedef _Float16  v8h  __attribute__((ext_vector_type(8)));
typedef _Float16  v4h  __attribute__((ext_vector_type(4)));

#define B_DIM   64
#define S_DIM   1024
#define D_DIM   64
#define NROWS   (B_DIM * S_DIM)        // rows per tensor = 65536
#define NMASK   (B_DIM * S_DIM)        // 65536 (B, Sy)

#define ROW_H   72                     // padded LDS row stride in halves (144 B)
#define TILE_H  (16 * ROW_H)           // one 16x64 tile in LDS (halves)

// ---------------------------------------------------------------------------
// Phase 1: row L2-normalize fp32 [rows, 64] -> f16 [rows, 64]
// 16 lanes per row (float4 each), 2 rows per wave, 16 rows per 256-thr block.
// ---------------------------------------------------------------------------
__global__ void normalize_f16_kernel(const float* __restrict__ src,
                                     _Float16* __restrict__ dst) {
    const int lane = threadIdx.x & 31;
    const int wave = threadIdx.x >> 5;
    const int sub  = lane >> 4;
    const int l16  = lane & 15;
    const size_t row = (((size_t)blockIdx.x * 8 + wave) << 1) + sub;

    const float4 v = *(const float4*)(src + row * D_DIM + l16 * 4);
    float ss = v.x * v.x + v.y * v.y + v.z * v.z + v.w * v.w;
    ss += __shfl_xor(ss, 1, 32);
    ss += __shfl_xor(ss, 2, 32);
    ss += __shfl_xor(ss, 4, 32);
    ss += __shfl_xor(ss, 8, 32);
    const float inv = 1.0f / fmaxf(sqrtf(ss), 1e-8f);

    v4h o;
    o[0] = (_Float16)(v.x * inv);
    o[1] = (_Float16)(v.y * inv);
    o[2] = (_Float16)(v.z * inv);
    o[3] = (_Float16)(v.w * inv);
    *(v4h*)(dst + row * D_DIM + l16 * 4) = o;
}

// ---------------------------------------------------------------------------
// Fragment loaders (wave32 WMMA layouts, 16-bit, 16x16x32)
// A (16x32, MxK): lane = 16*half + m holds row m;
//   elems 0..7  -> K = 8*half + [0..7] ; elems 8..15 -> K = 16 + 8*half + [0..7]
// B (32x16, KxN): lane = 16*half + n holds col n; elems same K mapping as A
// rowk points at the row's first K element, so the same loader works for global and LDS rows.
// ---------------------------------------------------------------------------
__device__ __forceinline__ v16h load_a_frag(const _Float16* rowk, int half) {
    const v8h lo = *(const v8h*)(rowk + 8 * half);
    const v8h hi = *(const v8h*)(rowk + 16 + 8 * half);
    v16h r;
#pragma unroll
    for (int i = 0; i < 8; ++i) { r[i] = lo[i]; r[i + 8] = hi[i]; }
    return r;
}

__device__ __forceinline__ v16h load_b_frag(const _Float16* rowk, int half) {
    const _Float16* p = rowk + 8 * half;
    const v8h lo = *(const v8h*)(p);
    const v8h hi = *(const v8h*)(p + 16);
    v16h r;
#pragma unroll
    for (int i = 0; i < 8; ++i) { r[i] = lo[i]; r[i + 8] = hi[i]; }
    return r;
}

// ---------------------------------------------------------------------------
// Async DMA of one 16x64 f16 A-tile (2048 B) global -> LDS, 128 lanes x b128.
// Uniform per-wave predicate (waves 0-3 issue, 4-7 idle) keeps EXEC full
// elsewhere. Tracked by ASYNCcnt.
// ---------------------------------------------------------------------------
__device__ __forceinline__ void async_tile_load(const _Float16* gsrc,
                                                unsigned lds_base, int tid) {
    if (tid < 128) {
        const unsigned lds =
            lds_base + (unsigned)(((tid >> 3) * ROW_H + (tid & 7) * 8) *
                                  (int)sizeof(_Float16));
        const unsigned long long g =
            (unsigned long long)(uintptr_t)gsrc + (unsigned)(tid * 16);
        asm volatile("global_load_async_to_lds_b128 %0, %1, off"
                     :: "v"(lds), "v"(g)
                     : "memory");
    }
}

__device__ __forceinline__ void wait_async_and_barrier() {
    asm volatile("s_wait_asynccnt 0x0" ::: "memory");
    __syncthreads();
}

// ---------------------------------------------------------------------------
// Phase 2: per (b, pair of 16-wide y-tiles): max over all Sx of pn.yn
// Block = 8 waves, one batch b, 16 y-tiles. A-tiles staged in LDS via async
// DMA (double buffered); each wave owns 2 y-tiles -> 4 WMMAs per A fragment.
// ---------------------------------------------------------------------------
__global__ void simmax_kernel(const _Float16* __restrict__ predh,
                              const _Float16* __restrict__ yh,
                              float* __restrict__ maxbuf) {
    __shared__ __align__(16) _Float16 smem[2 * TILE_H];

    const int tid  = threadIdx.x;
    const int lane = tid & 31;
    const int wave = tid >> 5;
    const int b    = blockIdx.x >> 2;         // 4 blocks per batch
    const int grp  = blockIdx.x & 3;          // 16 y-tiles per block
    const int yt0  = grp * 16 + wave * 2;
    const int yt1  = yt0 + 1;
    const int half = lane >> 4;
    const int mn   = lane & 15;

    // B fragments for both y-tiles (K split 0..31 / 32..63), kept in VGPRs
    const _Float16* yb0 =
        yh + (((size_t)b * S_DIM) + (size_t)yt0 * 16 + mn) * D_DIM;
    const _Float16* yb1 =
        yh + (((size_t)b * S_DIM) + (size_t)yt1 * 16 + mn) * D_DIM;
    const v16h b00 = load_b_frag(yb0, half);
    const v16h b01 = load_b_frag(yb0 + 32, half);
    const v16h b10 = load_b_frag(yb1, half);
    const v16h b11 = load_b_frag(yb1 + 32, half);

    const _Float16* pbase = predh + (size_t)b * S_DIM * D_DIM;
    const unsigned lds0 = (unsigned)(uintptr_t)&smem[0];
    const unsigned ldsN = (unsigned)(TILE_H * sizeof(_Float16));

    // Preload tile 0
    async_tile_load(pbase, lds0, tid);
    wait_async_and_barrier();

    float best0 = -2.0f, best1 = -2.0f;
    for (int xt = 0; xt < 64; ++xt) {
        const int cur = xt & 1;
        if (xt + 1 < 64)   // prefetch next tile into the other buffer
            async_tile_load(pbase + (size_t)(xt + 1) * 16 * D_DIM,
                            lds0 + (cur ^ 1) * ldsN, tid);

        const _Float16* arow = &smem[cur * TILE_H + mn * ROW_H];
        const v16h a0 = load_a_frag(arow, half);       // ds_load_b128 x2
        const v16h a1 = load_a_frag(arow + 32, half);  // ds_load_b128 x2

        v8f c0 = {}, c1 = {};
        c0 = __builtin_amdgcn_wmma_f32_16x16x32_f16(false, a0, false, b00,
                                                    (short)0, c0, false, false);
        c0 = __builtin_amdgcn_wmma_f32_16x16x32_f16(false, a1, false, b01,
                                                    (short)0, c0, false, false);
        c1 = __builtin_amdgcn_wmma_f32_16x16x32_f16(false, a0, false, b10,
                                                    (short)0, c1, false, false);
        c1 = __builtin_amdgcn_wmma_f32_16x16x32_f16(false, a1, false, b11,
                                                    (short)0, c1, false, false);
#pragma unroll
        for (int j = 0; j < 8; ++j) {
            best0 = fmaxf(best0, c0[j]);
            best1 = fmaxf(best1, c1[j]);
        }
        wait_async_and_barrier();   // next tile landed; all reads of cur done
    }

    // lanes 0-15 hold max over M=0..7 (N=lane), lanes 16-31 over M=8..15
    best0 = fmaxf(best0, __shfl_xor(best0, 16, 32));
    best1 = fmaxf(best1, __shfl_xor(best1, 16, 32));
    if (lane < 16) {
        maxbuf[((size_t)b * S_DIM) + (size_t)yt0 * 16 + lane] = best0;
        maxbuf[((size_t)b * S_DIM) + (size_t)yt1 * 16 + lane] = best1;
    }
}

// ---------------------------------------------------------------------------
// Phase 3: deterministic two-stage masked mean of (1 - max_sim)
// ---------------------------------------------------------------------------
__global__ void reduce_partial_kernel(const float* __restrict__ maxbuf,
                                      const int* __restrict__ y_mask,
                                      float* __restrict__ pL,
                                      float* __restrict__ pM) {
    __shared__ float sL[256];
    __shared__ float sM[256];
    float aL = 0.0f, aM = 0.0f;
    for (int i = blockIdx.x * 256 + threadIdx.x; i < NMASK; i += 64 * 256) {
        const float m = (float)y_mask[i];
        aL += (1.0f - maxbuf[i]) * m;
        aM += m;
    }
    sL[threadIdx.x] = aL;
    sM[threadIdx.x] = aM;
    __syncthreads();
#pragma unroll
    for (int s = 128; s > 0; s >>= 1) {
        if (threadIdx.x < s) {
            sL[threadIdx.x] += sL[threadIdx.x + s];
            sM[threadIdx.x] += sM[threadIdx.x + s];
        }
        __syncthreads();
    }
    if (threadIdx.x == 0) {
        pL[blockIdx.x] = sL[0];
        pM[blockIdx.x] = sM[0];
    }
}

__global__ void reduce_final_kernel(const float* __restrict__ pL,
                                    const float* __restrict__ pM,
                                    float* __restrict__ out) {
    __shared__ float sL[64];
    __shared__ float sM[64];
    const int t = threadIdx.x;
    sL[t] = pL[t];
    sM[t] = pM[t];
    __syncthreads();
#pragma unroll
    for (int s = 32; s > 0; s >>= 1) {
        if (t < s) { sL[t] += sL[t + s]; sM[t] += sM[t + s]; }
        __syncthreads();
    }
    if (t == 0) out[0] = sL[0] / sM[0];
}

// ---------------------------------------------------------------------------
extern "C" void kernel_launch(void* const* d_in, const int* in_sizes, int n_in,
                              void* d_out, int out_size, void* d_ws,
                              size_t ws_size, hipStream_t stream) {
    const float* pred   = (const float*)d_in[0];  // [64,1024,64] fp32
    const float* y      = (const float*)d_in[1];  // [64,1024,64] fp32
    const int*   y_mask = (const int*)d_in[2];    // [64,1024] int32
    float* out = (float*)d_out;                   // scalar fp32

    char* ws = (char*)d_ws;
    const size_t half_bytes = (size_t)NROWS * D_DIM * sizeof(_Float16); // 8 MB
    _Float16* predh  = (_Float16*)ws;
    _Float16* yh     = (_Float16*)(ws + half_bytes);
    float*    maxbuf = (float*)(ws + 2 * half_bytes);                   // 256 KB
    float*    pL     = (float*)(ws + 2 * half_bytes + (size_t)NMASK * 4);
    float*    pM     = pL + 64;

    // Phase 1: normalize + fp32->f16
    normalize_f16_kernel<<<NROWS / 16, 256, 0, stream>>>(pred, predh);
    normalize_f16_kernel<<<NROWS / 16, 256, 0, stream>>>(y, yh);

    // Phase 2: WMMA sim + fused max. 256 blocks x 8 waves, 2 y-tiles/wave,
    // A-tiles async-DMA'd into double-buffered LDS.
    simmax_kernel<<<256, 256, 0, stream>>>(predh, yh, maxbuf);

    // Phase 3: masked mean (deterministic)
    reduce_partial_kernel<<<64, 256, 0, stream>>>(maxbuf, y_mask, pL, pM);
    reduce_final_kernel<<<1, 64, 0, stream>>>(pL, pM, out);
}
